// NTXentLoss_70188355551618
// MI455X (gfx1250) — matrix operand order repaired and checked by reading, not compiled, not run on hardware
//
#include <hip/hip_runtime.h>
#include <hip/hip_bf16.h>

typedef __attribute__((ext_vector_type(16))) _Float16 v16h;
typedef __attribute__((ext_vector_type(8)))  float    v8f;

#define BATCH 4096
#define NROWS 8192              // 2B
#define DIM   256
#define TILE  128               // column tile (rows of zn staged in LDS)
#define NTILES (NROWS / TILE)   // 64
#define CGROUPS 4               // column split: 4 groups of 16 tiles
#define TILES_PER_CG (NTILES / CGROUPS)   // 16
#define TILE_BYTES (TILE * DIM * 2)       // 65536
#define TWO_OVER_LN2 2.8853900817779268f  // exp(2x) = exp2(x * 2/ln2)

union F16x16 { v16h v; uint4 q[2]; };
union H4     { _Float16 h[4]; uint2 u; };
union H8     { _Float16 h[8]; uint4 q; };

// ---------------------------------------------------------------------------
// Kernel 1: row-normalize z = [z_i; z_j] (f32) -> zn (f16, row-major)
// ---------------------------------------------------------------------------
__global__ __launch_bounds__(256)
void ntxent_normalize(const float* __restrict__ z_i,
                      const float* __restrict__ z_j,
                      _Float16* __restrict__ zn)
{
    const int wave = threadIdx.x >> 5;
    const int lane = threadIdx.x & 31;
    const int row  = blockIdx.x * 8 + wave;          // 0..8191

    const float* src = (row < BATCH) ? (z_i + (size_t)row * DIM)
                                     : (z_j + (size_t)(row - BATCH) * DIM);

    const float4 a = ((const float4*)src)[lane];
    const float4 b = ((const float4*)src)[lane + 32];

    float ss = a.x*a.x + a.y*a.y + a.z*a.z + a.w*a.w
             + b.x*b.x + b.y*b.y + b.z*b.z + b.w*b.w;
    #pragma unroll
    for (int off = 16; off >= 1; off >>= 1)
        ss += __shfl_xor(ss, off, 32);

    const float inv = 1.0f / fmaxf(sqrtf(ss), 1e-8f);

    _Float16* dst = zn + (size_t)row * DIM;
    H4 p0, p1;
    p0.h[0] = (_Float16)(a.x * inv); p0.h[1] = (_Float16)(a.y * inv);
    p0.h[2] = (_Float16)(a.z * inv); p0.h[3] = (_Float16)(a.w * inv);
    p1.h[0] = (_Float16)(b.x * inv); p1.h[1] = (_Float16)(b.y * inv);
    p1.h[2] = (_Float16)(b.z * inv); p1.h[3] = (_Float16)(b.w * inv);
    ((uint2*)dst)[lane]      = p0.u;
    ((uint2*)dst)[lane + 32] = p1.u;
}

// ---------------------------------------------------------------------------
// Kernel 1b: positive-pair logits (f32 dot over the f16 normalized data)
// ---------------------------------------------------------------------------
__global__ __launch_bounds__(256)
void ntxent_pos(const _Float16* __restrict__ zn, float* __restrict__ poslogit)
{
    const int wave = threadIdx.x >> 5;
    const int lane = threadIdx.x & 31;
    const int i = blockIdx.x * 8 + wave;             // 0..4095

    H8 ua, ub;
    ua.q = ((const uint4*)(zn + (size_t)i * DIM))[lane];
    ub.q = ((const uint4*)(zn + (size_t)(i + BATCH) * DIM))[lane];

    float acc = 0.0f;
    #pragma unroll
    for (int j = 0; j < 8; ++j)
        acc += (float)ua.h[j] * (float)ub.h[j];
    #pragma unroll
    for (int off = 16; off >= 1; off >>= 1)
        acc += __shfl_xor(acc, off, 32);

    if (lane == 0) {
        const float s = acc + acc;                   // dot / TEMPERATURE
        poslogit[i]         = s;
        poslogit[i + BATCH] = s;
    }
}

// ---------------------------------------------------------------------------
// Kernel 2: Gram partial row-sums via v_wmma_f32_16x16x32_f16.
// Grid (64 row-blocks x 4 column-groups) = 256 WGs so the whole chip is busy.
// Each WG: 128 rows x 16 column tiles; 8 waves, A resident in VGPRs;
// tiles double-buffered in 2x64KB LDS via global_load_async_to_lds_b128.
// Partials written to rowsum4[cg][row]; diagonal term subtracted only by the
// column-group that owns the diagonal tile (uniform condition).
// ---------------------------------------------------------------------------
__global__ __launch_bounds__(256)
void ntxent_gram_rowsum(const _Float16* __restrict__ zn,
                        float* __restrict__ rowsum4 /* [CGROUPS][NROWS] */)
{
    __shared__ _Float16 lds[2 * TILE * DIM];         // 128 KB (2 WGs / WGP)

    const int wave   = threadIdx.x >> 5;
    const int lane   = threadIdx.x & 31;
    const int hi     = lane >> 4;
    const int rowBlk = blockIdx.x;                   // 0..63
    const int cg     = blockIdx.y;                   // 0..3
    const int rowBase = rowBlk * 128 + wave * 16;
    const int t0      = cg * TILES_PER_CG;           // first column tile

    // --- resident A fragments: ISA 16-bit A 16x32 layout ---
    F16x16 A[8];
    {
        const int m = rowBase + (lane & 15);
        const size_t base = (size_t)m * DIM + hi * 8;
        #pragma unroll
        for (int kc = 0; kc < 8; ++kc) {
            const _Float16* p = zn + base + kc * 32;
            A[kc].q[0] = *(const uint4*)(p);
            A[kc].q[1] = *(const uint4*)(p + 16);
        }
    }

    // row self-dot for the diagonal correction (lanes l, l^16 hold the two
    // K-halves of row lane&15)
    float ssq = 0.0f;
    #pragma unroll
    for (int kc = 0; kc < 8; ++kc)
        #pragma unroll
        for (int j = 0; j < 16; ++j) {
            const float h = (float)A[kc].v[j];
            ssq += h * h;
        }
    const float full_sd = ssq + __shfl_xor(ssq, 16, 32);

    // async copy of one 64KB column tile: 256 threads x 16 x 16B
    const char* gbase = (const char*)zn;
    const unsigned ldsBase = (unsigned)(uintptr_t)&lds[0] + threadIdx.x * 16u;
    auto issueTile = [&](int tt, int buf) {
        const char* g = gbase + (size_t)tt * TILE_BYTES + threadIdx.x * 16u;
        const unsigned l = ldsBase + (unsigned)buf * TILE_BYTES;
        #pragma unroll
        for (int i = 0; i < 16; ++i) {
            asm volatile("global_load_async_to_lds_b128 %0, %1, off"
                         :: "v"(l + i * 4096u), "v"(g + (size_t)i * 4096)
                         : "memory");
        }
    };

    float rs[8];
    #pragma unroll
    for (int r = 0; r < 8; ++r) rs[r] = 0.0f;

    issueTile(t0, 0);

    for (int tt = 0; tt < TILES_PER_CG; ++tt) {
        asm volatile("s_wait_asynccnt 0" ::: "memory");
        __syncthreads();                 // tile tt in LDS; compute(tt-1) done
        if (tt + 1 < TILES_PER_CG)
            issueTile(t0 + tt + 1, (tt + 1) & 1);   // overlaps compute below

        const _Float16* cur = lds + (size_t)(tt & 1) * (TILE * DIM);

        #pragma unroll 2
        for (int nt = 0; nt < 8; ++nt) {
            // B fragment: ISA 16-bit B 32x16 layout; explicit double buffer
            const int n_local = nt * 16 + (lane & 15);
            const _Float16* brow = cur + (size_t)n_local * DIM + hi * 8;

            F16x16 Bf[2];
            {
                const uint4* p = (const uint4*)brow;
                Bf[0].q[0] = p[0];
                Bf[0].q[1] = p[2];
            }
            v8f c = {};
            #pragma unroll
            for (int kc = 0; kc < 8; ++kc) {
                if (kc + 1 < 8) {
                    const uint4* p = (const uint4*)(brow + (kc + 1) * 32);
                    Bf[(kc + 1) & 1].q[0] = p[0];
                    Bf[(kc + 1) & 1].q[1] = p[2];
                }
                c = __builtin_amdgcn_wmma_f32_16x16x32_f16(
                        false, A[kc].v, false, Bf[kc & 1].v,
                        (short)0, c, false, false);
            }

            // C layout: lane holds (M = r + 8*hi, N = lane&15) in c[r]
            #pragma unroll
            for (int r = 0; r < 8; ++r)
                rs[r] += __builtin_amdgcn_exp2f(c[r] * TWO_OVER_LN2);
        }
    }

    // reduce partials across the 16 lanes of each half-wave; the column-group
    // owning the diagonal tile (tile index == rowBlk) removes exp(2*selfdot)
    const bool diagHere = ((rowBlk >> 4) == cg);
    #pragma unroll
    for (int r = 0; r < 8; ++r) {
        float v = rs[r];
        #pragma unroll
        for (int off = 1; off < 16; off <<= 1)
            v += __shfl_xor(v, off, 32);
        const float sd = __shfl(full_sd, 8 * hi + r, 32);   // row 8*hi+r
        if (diagHere)
            v -= __builtin_amdgcn_exp2f(sd * TWO_OVER_LN2);
        if ((lane & 15) == 0)
            rowsum4[(size_t)cg * NROWS + rowBase + 8 * hi + r] = v;
    }
}

// ---------------------------------------------------------------------------
// Kernel 3: loss = mean_i( log(sum_cg rowsum4[cg][i]) - poslogit[i] )
// deterministic 4-way partial sum, then block tree-reduce
// ---------------------------------------------------------------------------
__global__ __launch_bounds__(256)
void ntxent_reduce(const float* __restrict__ rowsum4,
                   const float* __restrict__ poslogit,
                   float* __restrict__ out)
{
    __shared__ float sbuf[256];
    float acc = 0.0f;
    for (int i = threadIdx.x; i < NROWS; i += 256) {
        const float r = rowsum4[i]
                      + rowsum4[NROWS + i]
                      + rowsum4[2 * NROWS + i]
                      + rowsum4[3 * NROWS + i];
        acc += logf(r) - poslogit[i];
    }
    sbuf[threadIdx.x] = acc;
    __syncthreads();
    #pragma unroll
    for (int s = 128; s > 0; s >>= 1) {
        if (threadIdx.x < s) sbuf[threadIdx.x] += sbuf[threadIdx.x + s];
        __syncthreads();
    }
    if (threadIdx.x == 0)
        out[0] = sbuf[0] / (float)NROWS;
}

// ---------------------------------------------------------------------------
extern "C" void kernel_launch(void* const* d_in, const int* in_sizes, int n_in,
                              void* d_out, int out_size, void* d_ws, size_t ws_size,
                              hipStream_t stream)
{
    const float* z_i = (const float*)d_in[0];
    const float* z_j = (const float*)d_in[1];
    float*       out = (float*)d_out;

    // ws: zn f16 [8192*256] (4MB) | rowsum4 f32 [4*8192] | poslogit f32 [8192]
    _Float16* zn       = (_Float16*)d_ws;
    float*    rowsum4  = (float*)((char*)d_ws + (size_t)NROWS * DIM * sizeof(_Float16));
    float*    poslogit = rowsum4 + CGROUPS * NROWS;

    ntxent_normalize<<<NROWS / 8, 256, 0, stream>>>(z_i, z_j, zn);
    ntxent_pos<<<BATCH / 8, 256, 0, stream>>>(zn, poslogit);
    ntxent_gram_rowsum<<<dim3(NROWS / 128, CGROUPS), 256, 0, stream>>>(zn, rowsum4);
    ntxent_reduce<<<1, 256, 0, stream>>>(rowsum4, poslogit, out);
}
